// WeakGNN_74594991997219
// MI455X (gfx1250) — matrix operand order repaired and checked by reading, not compiled, not run on hardware
//
#include <hip/hip_runtime.h>

#define N_NODES   100000
#define N_EDGES   1600000
#define D         64
#define N_LAYERS  4

typedef __attribute__((ext_vector_type(16))) _Float16 v16h;
typedef __attribute__((ext_vector_type(8)))  float    v8f;

// ---------------------------------------------------------------------------
// Weight prep: convert 7 [64x64] f32 matrices (row-major, K x N) into f16
// WMMA B-fragments. Fragment f = nt*2 + ks covers N-tile nt (16 cols) and
// K-step ks (32 rows).  Per ISA 7.12.2 (16-bit B 32x16): lane n<16 holds
// column n with K = ks*32 + 0..7 in halves 0..7 and 16..23 in halves 8..15;
// lane 16+n holds column n with K = ks*32 + 8..15, 24..31.  Stored as [mat][frag][lane][16 halves].
// ---------------------------------------------------------------------------
__global__ __launch_bounds__(256)
void prep_weights(const float* __restrict__ W_ne, const float* __restrict__ W_ee,
                  const float* __restrict__ W_conv, const float* __restrict__ W_lin,
                  _Float16* __restrict__ wswz)
{
    int idx = blockIdx.x * 256 + threadIdx.x;          // 7 * 4096 elements
    if (idx >= 7 * 4096) return;
    int mat  = idx >> 12;
    int r    = idx & 4095;
    int f    = r >> 9;          // 0..7
    int lane = (r >> 4) & 31;
    int i    = r & 15;
    int nt = f >> 1, ks = f & 1;
    int K = ks * 32 + (lane >> 4) * 8 + (i & 7) + (i >> 3) * 16;
    int N = nt * 16 + (lane & 15);
    const float* src;
    if      (mat == 0) src = W_ne;
    else if (mat == 1) src = W_ee;
    else if (mat <= 5) src = W_conv + (size_t)(mat - 2) * 4096;
    else               src = W_lin;
    wswz[idx] = (_Float16)src[K * 64 + N];
}

// ---------------------------------------------------------------------------
// WMMA GEMM:  C[M,64] = op(A)[M,64] @ W[64,64] + bias (+resid)
//   FUSE_AGGR:  op(A) = A + numer/(denom+1e-16)   (GENConv aggregation)
//   RESID:      C += resid                         (pre-norm residual)
//   OUT_F16:    store C as f16                     (edge encoder)
// One wave per 16-row tile; 8 waves per block.  M must be a multiple of 16.
// A-fragment layout (16-bit A 16x32): lane m (m<16) holds row m, halves
// 0..7 = K klo..klo+7, halves 8..15 = K 16+klo..23+klo with klo=(lane>>4)*8.
// ---------------------------------------------------------------------------
template<bool FUSE_AGGR, bool RESID, bool OUT_F16>
__global__ __launch_bounds__(256)
void gemm64(const float* __restrict__ A,
            const float* __restrict__ numer, const float* __restrict__ denom,
            const _Float16* __restrict__ Wswz, const float* __restrict__ bias,
            const float* __restrict__ resid,
            float* __restrict__ Cf32, _Float16* __restrict__ Cf16, int M)
{
    int lane = threadIdx.x & 31;
    int wave = threadIdx.x >> 5;
    long tile   = (long)blockIdx.x * 8 + wave;
    long ntiles = (long)(M >> 4);
    if (tile >= ntiles) return;                 // wave-uniform: EXEC stays full

    int  mrow = lane & 15;
    int  klo  = (lane >> 4) * 8;
    long row  = tile * 16 + mrow;
    const float* arow = A + row * 64;

    v16h a0, a1;
#pragma unroll
    for (int i = 0; i < 8; ++i) {
        float x0 = arow[klo + i];
        float x1 = arow[16 + klo + i];
        float x2 = arow[32 + klo + i];
        float x3 = arow[48 + klo + i];
        if (FUSE_AGGR) {
            long b = row * 64;
            x0 += numer[b + klo + i]      / (denom[b + klo + i]      + 1e-16f);
            x1 += numer[b + 16 + klo + i] / (denom[b + 16 + klo + i] + 1e-16f);
            x2 += numer[b + 32 + klo + i] / (denom[b + 32 + klo + i] + 1e-16f);
            x3 += numer[b + 48 + klo + i] / (denom[b + 48 + klo + i] + 1e-16f);
        }
        a0[i] = (_Float16)x0; a0[i + 8] = (_Float16)x1;
        a1[i] = (_Float16)x2; a1[i + 8] = (_Float16)x3;
    }

#pragma unroll
    for (int nt = 0; nt < 4; ++nt) {
        v16h b0 = *(const v16h*)(Wswz + (nt * 2 + 0) * 512 + lane * 16);
        v16h b1 = *(const v16h*)(Wswz + (nt * 2 + 1) * 512 + lane * 16);
        v8f c = {};
        c = __builtin_amdgcn_wmma_f32_16x16x32_f16(false, a0, false, b0,
                                                   (short)0, c, false, false);
        c = __builtin_amdgcn_wmma_f32_16x16x32_f16(false, a1, false, b1,
                                                   (short)0, c, false, false);
        float bv = bias[nt * 16 + (lane & 15)];
#pragma unroll
        for (int r = 0; r < 8; ++r) {
            long orow = tile * 16 + r + (lane >> 4) * 8;   // C layout: lanes>=16 -> M+8
            long oidx = orow * 64 + nt * 16 + (lane & 15);
            float v = c[r] + bv;
            if (RESID) v += resid[oidx];
            if (OUT_F16) Cf16[oidx] = (_Float16)v;
            else         Cf32[oidx] = v;
        }
    }
}

// ---------------------------------------------------------------------------
// Edge pass 1: channel-wise segment max.  All w > 0 (msg >= 1e-7, t > 0),
// so uint-bit atomicMax == float max; m buffer is pre-zeroed (0 bits == 0.0f).
// ---------------------------------------------------------------------------
__global__ __launch_bounds__(256)
void edge_max_k(const float* __restrict__ z, const _Float16* __restrict__ e16,
                const int* __restrict__ ei, const float* __restrict__ tptr,
                int layer, unsigned* __restrict__ mbits)
{
    int idx = blockIdx.x * 256 + threadIdx.x;           // E*64 = 102.4M < 2^31
    if (idx >= N_EDGES * D) return;
    int edge = idx >> 6, c = idx & 63;
    int src = ei[edge];
    int dst = ei[N_EDGES + edge];
    float t   = tptr[layer];
    float msg = fmaxf(z[src * 64 + c] + (float)e16[idx], 0.0f) + 1e-7f;
    float w   = msg * t;
    atomicMax(&mbits[dst * 64 + c], __float_as_uint(w));
}

// Edge pass 2: ex = exp(w - m[dst]); accumulate denom += ex, numer += ex*msg.
__global__ __launch_bounds__(256)
void edge_sum_k(const float* __restrict__ z, const _Float16* __restrict__ e16,
                const int* __restrict__ ei, const float* __restrict__ tptr,
                int layer, const float* __restrict__ m,
                float* __restrict__ denom, float* __restrict__ numer)
{
    int idx = blockIdx.x * 256 + threadIdx.x;
    if (idx >= N_EDGES * D) return;
    int edge = idx >> 6, c = idx & 63;
    int src = ei[edge];
    int dst = ei[N_EDGES + edge];
    float t   = tptr[layer];
    float msg = fmaxf(z[src * 64 + c] + (float)e16[idx], 0.0f) + 1e-7f;
    float w   = msg * t;
    int   dc  = dst * 64 + c;
    float ex  = __expf(w - m[dc]);
    atomicAdd(&denom[dc], ex);
    atomicAdd(&numer[dc], ex * msg);
}

// ---------------------------------------------------------------------------
// Wave-per-node LayerNorm + ReLU (lane handles channels lane, lane+32).
// ---------------------------------------------------------------------------
__global__ __launch_bounds__(256)
void ln_relu(const float* __restrict__ h, const float* __restrict__ gamma,
             const float* __restrict__ beta, float* __restrict__ z, int N)
{
    int lane = threadIdx.x & 31;
    int wave = threadIdx.x >> 5;
    int node = blockIdx.x * 8 + wave;
    if (node >= N) return;
    float v0 = h[node * 64 + lane];
    float v1 = h[node * 64 + 32 + lane];
    float s = v0 + v1;
#pragma unroll
    for (int o = 16; o >= 1; o >>= 1) s += __shfl_xor(s, o, 32);
    float mu = s * (1.0f / 64.0f);
    float d0 = v0 - mu, d1 = v1 - mu;
    float q = d0 * d0 + d1 * d1;
#pragma unroll
    for (int o = 16; o >= 1; o >>= 1) q += __shfl_xor(q, o, 32);
    float rinv = rsqrtf(q * (1.0f / 64.0f) + 1e-5f);
    z[node * 64 + lane]      = fmaxf(d0 * rinv * gamma[lane]      + beta[lane],      0.0f);
    z[node * 64 + 32 + lane] = fmaxf(d1 * rinv * gamma[32 + lane] + beta[32 + lane], 0.0f);
}

// Zero m/denom/numer (contiguous 3 * N_NODES * 64 floats) with float4 stores.
__global__ __launch_bounds__(256)
void zero3_k(float4* __restrict__ p)
{
    int idx = blockIdx.x * 256 + threadIdx.x;           // 4.8M float4
    if (idx >= 3 * N_NODES * D / 4) return;
    p[idx] = make_float4(0.f, 0.f, 0.f, 0.f);
}

// ---------------------------------------------------------------------------
extern "C" void kernel_launch(void* const* d_in, const int* in_sizes, int n_in,
                              void* d_out, int out_size, void* d_ws, size_t ws_size,
                              hipStream_t stream)
{
    const float* x      = (const float*)d_in[0];
    const float* eattr  = (const float*)d_in[1];
    const int*   eidx   = (const int*)  d_in[2];
    const float* W_ne   = (const float*)d_in[3];
    const float* b_ne   = (const float*)d_in[4];
    const float* W_ee   = (const float*)d_in[5];
    const float* b_ee   = (const float*)d_in[6];
    const float* W_conv = (const float*)d_in[7];
    const float* b_conv = (const float*)d_in[8];
    const float* t      = (const float*)d_in[9];
    const float* gamma  = (const float*)d_in[10];
    const float* beta   = (const float*)d_in[11];
    const float* W_lin  = (const float*)d_in[12];
    const float* b_lin  = (const float*)d_in[13];
    float*       out    = (float*)d_out;

    const size_t NODE_F = (size_t)N_NODES * D;          // 6.4M floats (25.6MB)
    char* ws = (char*)d_ws;
    _Float16* wswz = (_Float16*)ws;                     // 7*4096 halves
    float* hA   = (float*)(ws + (1 << 16));
    float* hB   = hA   + NODE_F;
    float* zbuf = hB   + NODE_F;
    float* mbuf = zbuf + NODE_F;                        // m, denom, numer contiguous
    float* dbuf = mbuf + NODE_F;
    float* nbuf = dbuf + NODE_F;
    _Float16* e16 = (_Float16*)(nbuf + NODE_F);         // 1.6M*64 f16 (204.8MB)

    const int NODE_TILES = N_NODES / 16;                // 6250
    const int EDGE_TILES = N_EDGES / 16;                // 100000
    const int GB_NODE = (NODE_TILES + 7) / 8;           // 782
    const int GB_EDGE = EDGE_TILES / 8;                 // 12500
    const int GB_EC   = (N_EDGES * D) / 256;            // 400000
    const int GB_LN   = (N_NODES + 7) / 8;              // 12500
    const int GB_Z3   = (3 * N_NODES * D / 4) / 256;    // 18750

    prep_weights<<<112, 256, 0, stream>>>(W_ne, W_ee, W_conv, W_lin, wswz);

    // encoders
    gemm64<false, false, false><<<GB_NODE, 256, 0, stream>>>(
        x, nullptr, nullptr, wswz + 0 * 4096, b_ne, nullptr, hA, nullptr, N_NODES);
    gemm64<false, false, true><<<GB_EDGE, 256, 0, stream>>>(
        eattr, nullptr, nullptr, wswz + 1 * 4096, b_ee, nullptr, nullptr, e16, N_EDGES);

    float* h = hA; float* hn = hB;
    for (int layer = 0; layer < N_LAYERS; ++layer) {
        const float* zin = h;
        if (layer > 0) {
            ln_relu<<<GB_LN, 256, 0, stream>>>(h, gamma + layer * 64, beta + layer * 64,
                                               zbuf, N_NODES);
            zin = zbuf;
        }
        zero3_k<<<GB_Z3, 256, 0, stream>>>((float4*)mbuf);
        edge_max_k<<<GB_EC, 256, 0, stream>>>(zin, e16, eidx, t, layer, (unsigned*)mbuf);
        edge_sum_k<<<GB_EC, 256, 0, stream>>>(zin, e16, eidx, t, layer, mbuf, dbuf, nbuf);
        if (layer == 0)
            gemm64<true, false, false><<<GB_NODE, 256, 0, stream>>>(
                zin, nbuf, dbuf, wswz + 2 * 4096, b_conv, nullptr, hn, nullptr, N_NODES);
        else
            gemm64<true, true, false><<<GB_NODE, 256, 0, stream>>>(
                zin, nbuf, dbuf, wswz + (2 + layer) * 4096, b_conv + layer * 64,
                h, hn, nullptr, N_NODES);
        float* tmp = h; h = hn; hn = tmp;
    }

    // final pre-norm + linear
    ln_relu<<<GB_LN, 256, 0, stream>>>(h, gamma, beta, zbuf, N_NODES);
    gemm64<false, false, false><<<GB_NODE, 256, 0, stream>>>(
        zbuf, nullptr, nullptr, wswz + 6 * 4096, b_lin, nullptr, out, nullptr, N_NODES);
}
